// SimpleRNN_78632261255389
// MI455X (gfx1250) — matrix operand order repaired and checked by reading, not compiled, not run on hardware
//
#include <hip/hip_runtime.h>
#include <hip/hip_bf16.h>

// Problem constants (match reference): B,T,I,H,L,O
#define RNN_B 128
#define RNN_T 512
#define RNN_I 512
#define RNN_H 1024
#define RNN_L 4
#define RNN_O 512

typedef __attribute__((ext_vector_type(16))) __bf16 bf16x16;
typedef __attribute__((ext_vector_type(8)))  __bf16 bf16x8;
typedef __attribute__((ext_vector_type(8)))  float  f32x8;

union Frag16 { bf16x16 v; bf16x8 h[2]; };

// ---------------------------------------------------------------- utilities

__global__ void zero_bf16_kernel(__bf16* __restrict__ p, int n) {
  int i = blockIdx.x * blockDim.x + threadIdx.x;
  int stride = gridDim.x * blockDim.x;
  for (; i < n; i += stride) p[i] = (__bf16)0.0f;
}

__global__ void cvt_f32_bf16_kernel(const float* __restrict__ in,
                                    __bf16* __restrict__ out, int n) {
  int i = blockIdx.x * blockDim.x + threadIdx.x;
  int stride = gridDim.x * blockDim.x;
  for (; i < n; i += stride) out[i] = (__bf16)in[i];
}

// ------------------------------------------------------- fragment loaders
// A (16x32 bf16, M x K): low lanes hold K {k0..k0+7, k0+16..k0+23}, high +8
__device__ __forceinline__ bf16x16 load_frag_a(const __bf16* row, int k0, int half) {
  Frag16 f;
  f.h[0] = *(const bf16x8*)(row + k0 + 8 * half);
  f.h[1] = *(const bf16x8*)(row + k0 + 16 + 8 * half);
  return f.v;
}
// B (32x16 bf16, K x N): low lanes hold K {k0..k0+7, k0+16..k0+23}, high +8
__device__ __forceinline__ bf16x16 load_frag_b(const __bf16* row, int k0, int half) {
  Frag16 f;
  f.h[0] = *(const bf16x8*)(row + k0 + 8 * half);
  f.h[1] = *(const bf16x8*)(row + k0 + 16 + 8 * half);
  return f.v;
}
// A fragment from fp32 source (layer-0 reads x directly, converts inline)
__device__ __forceinline__ bf16x16 load_frag_a_f32(const float* row, int k0, int half) {
  f32x8 f0 = *(const f32x8*)(row + k0 + 8 * half);
  f32x8 f1 = *(const f32x8*)(row + k0 + 16 + 8 * half);
  Frag16 f;
  #pragma unroll
  for (int i = 0; i < 8; ++i) {
    f.h[0][i] = (__bf16)f0[i];
    f.h[1][i] = (__bf16)f1[i];
  }
  return f.v;
}

#define WMMA_BF16(A, B, C) \
  __builtin_amdgcn_wmma_f32_16x16x32_bf16(false, (A), false, (B), (short)0, (C), false, false)

// ---------------------------------------------------------------- RNN layer
// Hnew = tanh(Ain @ Wih^T + bih + Hold @ Whh^T + bhh)
// Register-blocked 32x32 output tile per wave: 2 A-frags x 2 B-frags ->
// 4 independent WMMA chains per K-step (ILP + 2x less L2 traffic).
// Grid: 4 m-groups x 32 n-groups = 128 waves = 16 blocks x 256 threads.

template <bool AF32>
__global__ __launch_bounds__(256)
void rnn_layer_kernel(const void* __restrict__ Ain, int Ka, long aStride,
                      const __bf16* __restrict__ Hold,
                      const __bf16* __restrict__ Wih,
                      const __bf16* __restrict__ Whh,
                      const float* __restrict__ bih,
                      const float* __restrict__ bhh,
                      __bf16* __restrict__ Hnew) {
  const int lane   = threadIdx.x & 31;
  const int laneLo = lane & 15;
  const int half   = lane >> 4;
  const int gw     = blockIdx.x * 8 + (threadIdx.x >> 5);  // 0..127
  const int m0     = (gw & 3) * 32;                        // batch-row group
  const int n0     = (gw >> 2) * 32;                       // hidden-col group

  f32x8 acc00 = {0.f,0.f,0.f,0.f,0.f,0.f,0.f,0.f};
  f32x8 acc01 = acc00, acc10 = acc00, acc11 = acc00;

  const int rA0 = m0 + laneLo, rA1 = m0 + 16 + laneLo;     // activation rows
  const int rB0 = n0 + laneLo, rB1 = n0 + 16 + laneLo;     // weight rows

  // ---- input GEMM: Ain @ Wih^T ----
  {
    const __bf16* w0 = Wih + (size_t)rB0 * Ka;
    const __bf16* w1 = Wih + (size_t)rB1 * Ka;
    if (AF32) {
      const float* a0 = (const float*)Ain + (size_t)rA0 * aStride;
      const float* a1 = (const float*)Ain + (size_t)rA1 * aStride;
      for (int k0 = 0; k0 < Ka; k0 += 32) {
        bf16x16 av0 = load_frag_a_f32(a0, k0, half);
        bf16x16 av1 = load_frag_a_f32(a1, k0, half);
        bf16x16 bv0 = load_frag_b(w0, k0, half);
        bf16x16 bv1 = load_frag_b(w1, k0, half);
        acc00 = WMMA_BF16(av0, bv0, acc00);
        acc01 = WMMA_BF16(av0, bv1, acc01);
        acc10 = WMMA_BF16(av1, bv0, acc10);
        acc11 = WMMA_BF16(av1, bv1, acc11);
      }
    } else {
      const __bf16* a0 = (const __bf16*)Ain + (size_t)rA0 * aStride;
      const __bf16* a1 = (const __bf16*)Ain + (size_t)rA1 * aStride;
      for (int k0 = 0; k0 < Ka; k0 += 32) {
        bf16x16 av0 = load_frag_a(a0, k0, half);
        bf16x16 av1 = load_frag_a(a1, k0, half);
        bf16x16 bv0 = load_frag_b(w0, k0, half);
        bf16x16 bv1 = load_frag_b(w1, k0, half);
        acc00 = WMMA_BF16(av0, bv0, acc00);
        acc01 = WMMA_BF16(av0, bv1, acc01);
        acc10 = WMMA_BF16(av1, bv0, acc10);
        acc11 = WMMA_BF16(av1, bv1, acc11);
      }
    }
  }

  // ---- recurrent GEMM: Hold @ Whh^T ----
  {
    const __bf16* a0 = Hold + (size_t)rA0 * RNN_H;
    const __bf16* a1 = Hold + (size_t)rA1 * RNN_H;
    const __bf16* w0 = Whh + (size_t)rB0 * RNN_H;
    const __bf16* w1 = Whh + (size_t)rB1 * RNN_H;
    for (int k0 = 0; k0 < RNN_H; k0 += 32) {
      bf16x16 av0 = load_frag_a(a0, k0, half);
      bf16x16 av1 = load_frag_a(a1, k0, half);
      bf16x16 bv0 = load_frag_b(w0, k0, half);
      bf16x16 bv1 = load_frag_b(w1, k0, half);
      acc00 = WMMA_BF16(av0, bv0, acc00);
      acc01 = WMMA_BF16(av0, bv1, acc01);
      acc10 = WMMA_BF16(av1, bv0, acc10);
      acc11 = WMMA_BF16(av1, bv1, acc11);
    }
  }

  // ---- epilogue: bias + tanh, store bf16 hidden state ----
  // C/D layout: acc[r] -> row (tile_m + r + 8*half), col (tile_n + laneLo)
  const float bias0 = bih[n0 + laneLo] + bhh[n0 + laneLo];
  const float bias1 = bih[n0 + 16 + laneLo] + bhh[n0 + 16 + laneLo];
  #pragma unroll
  for (int r = 0; r < 8; ++r) {
    const int mA = m0 + r + 8 * half;
    const int mB = m0 + 16 + r + 8 * half;
    Hnew[(size_t)mA * RNN_H + n0 + laneLo]      = (__bf16)tanhf(acc00[r] + bias0);
    Hnew[(size_t)mA * RNN_H + n0 + 16 + laneLo] = (__bf16)tanhf(acc01[r] + bias1);
    Hnew[(size_t)mB * RNN_H + n0 + laneLo]      = (__bf16)tanhf(acc10[r] + bias0);
    Hnew[(size_t)mB * RNN_H + n0 + 16 + laneLo] = (__bf16)tanhf(acc11[r] + bias1);
  }
}

// ---------------------------------------------------------------- final FC
// out[(b*T + t)*O + n] = Htop[b] @ Wfc^T + bfc, 32x32 tile per wave.
// Grid: 4 m-groups x 16 n-groups = 64 waves = 8 blocks x 256 threads.

__global__ __launch_bounds__(256)
void fc_kernel(const __bf16* __restrict__ Htop,
               const __bf16* __restrict__ Wfc,
               const float* __restrict__ bfc,
               float* __restrict__ out, int t) {
  const int lane   = threadIdx.x & 31;
  const int laneLo = lane & 15;
  const int half   = lane >> 4;
  const int gw     = blockIdx.x * 8 + (threadIdx.x >> 5);  // 0..63
  const int m0     = (gw & 3) * 32;
  const int n0     = (gw >> 2) * 32;

  f32x8 acc00 = {0.f,0.f,0.f,0.f,0.f,0.f,0.f,0.f};
  f32x8 acc01 = acc00, acc10 = acc00, acc11 = acc00;

  const __bf16* a0 = Htop + (size_t)(m0 + laneLo) * RNN_H;
  const __bf16* a1 = Htop + (size_t)(m0 + 16 + laneLo) * RNN_H;
  const __bf16* w0 = Wfc + (size_t)(n0 + laneLo) * RNN_H;
  const __bf16* w1 = Wfc + (size_t)(n0 + 16 + laneLo) * RNN_H;

  for (int k0 = 0; k0 < RNN_H; k0 += 32) {
    bf16x16 av0 = load_frag_a(a0, k0, half);
    bf16x16 av1 = load_frag_a(a1, k0, half);
    bf16x16 bv0 = load_frag_b(w0, k0, half);
    bf16x16 bv1 = load_frag_b(w1, k0, half);
    acc00 = WMMA_BF16(av0, bv0, acc00);
    acc01 = WMMA_BF16(av0, bv1, acc01);
    acc10 = WMMA_BF16(av1, bv0, acc10);
    acc11 = WMMA_BF16(av1, bv1, acc11);
  }

  const float bias0 = bfc[n0 + laneLo];
  const float bias1 = bfc[n0 + 16 + laneLo];
  #pragma unroll
  for (int r = 0; r < 8; ++r) {
    const int bA = m0 + r + 8 * half;        // batch index, first m-tile
    const int bB = m0 + 16 + r + 8 * half;   // batch index, second m-tile
    out[((size_t)bA * RNN_T + t) * RNN_O + n0 + laneLo]      = acc00[r] + bias0;
    out[((size_t)bA * RNN_T + t) * RNN_O + n0 + 16 + laneLo] = acc01[r] + bias1;
    out[((size_t)bB * RNN_T + t) * RNN_O + n0 + laneLo]      = acc10[r] + bias0;
    out[((size_t)bB * RNN_T + t) * RNN_O + n0 + 16 + laneLo] = acc11[r] + bias1;
  }
}

// ---------------------------------------------------------------- launcher

extern "C" void kernel_launch(void* const* d_in, const int* in_sizes, int n_in,
                              void* d_out, int out_size, void* d_ws, size_t ws_size,
                              hipStream_t stream) {
  (void)in_sizes; (void)n_in; (void)out_size; (void)ws_size;

  const float* x     = (const float*)d_in[0];   // (B,T,I)
  const float* W_ih0 = (const float*)d_in[1];   // (H,I)
  const float* b_ih0 = (const float*)d_in[2];   // (H)
  const float* W_hh0 = (const float*)d_in[3];   // (H,H)
  const float* b_hh0 = (const float*)d_in[4];   // (H)
  const float* W_ih  = (const float*)d_in[5];   // (L-1,H,H)
  const float* b_ih  = (const float*)d_in[6];   // (L-1,H)
  const float* W_hh  = (const float*)d_in[7];   // (L-1,H,H)
  const float* b_hh  = (const float*)d_in[8];   // (L-1,H)
  const float* W_fc  = (const float*)d_in[9];   // (O,H)
  const float* b_fc  = (const float*)d_in[10];  // (O)
  float* out = (float*)d_out;

  // workspace carve-out (bf16 weights stay L2-resident: ~17 MB total)
  char* ws = (char*)d_ws;
  size_t off = 0;
  auto carve = [&](size_t bytes) -> char* {
    char* p = ws + off;
    off = (off + bytes + 255) & ~(size_t)255;
    return p;
  };
  __bf16* wih0 = (__bf16*)carve((size_t)RNN_H * RNN_I * 2);
  __bf16* whh0 = (__bf16*)carve((size_t)RNN_H * RNN_H * 2);
  __bf16* wih  = (__bf16*)carve((size_t)(RNN_L - 1) * RNN_H * RNN_H * 2);
  __bf16* whh  = (__bf16*)carve((size_t)(RNN_L - 1) * RNN_H * RNN_H * 2);
  __bf16* wfc  = (__bf16*)carve((size_t)RNN_O * RNN_H * 2);
  __bf16* h0   = (__bf16*)carve((size_t)RNN_L * RNN_B * RNN_H * 2);  // ping
  __bf16* h1   = (__bf16*)carve((size_t)RNN_L * RNN_B * RNN_H * 2);  // pong

  // convert weights fp32 -> bf16 (row-major preserved; B-matrix needs no transpose)
  cvt_f32_bf16_kernel<<<512, 256, 0, stream>>>(W_ih0, wih0, RNN_H * RNN_I);
  cvt_f32_bf16_kernel<<<512, 256, 0, stream>>>(W_hh0, whh0, RNN_H * RNN_H);
  cvt_f32_bf16_kernel<<<1024, 256, 0, stream>>>(W_ih, wih, (RNN_L - 1) * RNN_H * RNN_H);
  cvt_f32_bf16_kernel<<<1024, 256, 0, stream>>>(W_hh, whh, (RNN_L - 1) * RNN_H * RNN_H);
  cvt_f32_bf16_kernel<<<512, 256, 0, stream>>>(W_fc, wfc, RNN_O * RNN_H);
  // h(-1) = 0
  zero_bf16_kernel<<<512, 256, 0, stream>>>(h0, RNN_L * RNN_B * RNN_H);

  const size_t layerHB = (size_t)RNN_B * RNN_H;  // elements per layer state
  const size_t HH = (size_t)RNN_H * RNN_H;

  for (int t = 0; t < RNN_T; ++t) {
    __bf16* prev = (t & 1) ? h1 : h0;   // state from step t-1
    __bf16* cur  = (t & 1) ? h0 : h1;   // state being produced at step t

    // layer 0: fp32 x_t (strided rows) + bf16 recurrent
    rnn_layer_kernel<true><<<16, 256, 0, stream>>>(
        (const void*)(x + (size_t)t * RNN_I), RNN_I, (long)RNN_T * RNN_I,
        prev, wih0, whh0, b_ih0, b_hh0, cur);

    // layers 1..3
    for (int l = 1; l < RNN_L; ++l) {
      rnn_layer_kernel<false><<<16, 256, 0, stream>>>(
          (const void*)(cur + (size_t)(l - 1) * layerHB), RNN_H, (long)RNN_H,
          prev + (size_t)l * layerHB,
          wih + (size_t)(l - 1) * HH, whh + (size_t)(l - 1) * HH,
          b_ih + (size_t)(l - 1) * RNN_H, b_hh + (size_t)(l - 1) * RNN_H,
          cur + (size_t)l * layerHB);
    }

    // fused per-timestep FC of the top-layer state -> d_out rows (b*T + t)
    fc_kernel<<<8, 256, 0, stream>>>(
        cur + (size_t)(RNN_L - 1) * layerHB, wfc, b_fc, out, t);
  }
}
